// Kappa_9723805958421
// MI455X (gfx1250) — compile-verified, matrix operand order checked by reading, not run on hardware
//
#include <hip/hip_runtime.h>
#include <hip/hip_bf16.h>

typedef __attribute__((ext_vector_type(16))) _Float16 v16h;
typedef __attribute__((ext_vector_type(8)))  _Float16 v8h;
typedef __attribute__((ext_vector_type(8)))  float    v8f;

#define NB 8
#define NN 4096
#define ND 64
#define NK 20
#define BN_EPS 1.0e-3f
#define TOTROWS (NB * NN)   // 32768
#define FINF 3.0e38f

static __device__ __forceinline__ v8h ld8(const _Float16* p) { return *(const v8h*)p; }
static __device__ __forceinline__ v16h mk16(v8h lo, v8h hi) {
  return __builtin_shufflevector(lo, hi, 0,1,2,3,4,5,6,7,8,9,10,11,12,13,14,15);
}
static __device__ __forceinline__ v8f wmma16(v16h a, v16h b, v8f c) {
  return __builtin_amdgcn_wmma_f32_16x16x32_f16(false, a, false, b, (short)0, c, false, false);
}
static __device__ __forceinline__ void lds_fence() {
  asm volatile("s_wait_dscnt 0" ::: "memory");
  __builtin_amdgcn_wave_barrier();
}
// CDNA5 async LDS fill: 16B global -> LDS, tracked by ASYNCcnt (no VGPR round trip)
static __device__ __forceinline__ void async_b128(void* lds, const void* g) {
  asm volatile("global_load_async_to_lds_b128 %0, %1, off"
               :: "v"((unsigned int)(unsigned long long)lds),
                  "v"((unsigned long long)g)
               : "memory");
}
static __device__ __forceinline__ void wait_async0() {
  asm volatile("s_wait_asynccnt 0" ::: "memory");
}

// ---------------- K0a: x(f32) -> xh(f16), row squared norms -------------
__global__ __launch_bounds__(256) void k0a_convert(const float* __restrict__ x,
                                                   _Float16* __restrict__ xh,
                                                   float* __restrict__ sq) {
  int row = blockIdx.x * 256 + threadIdx.x;      // 0..32767
  const float* xr = x + (size_t)row * ND;
  _Float16* hr = xh + (size_t)row * ND;
  float s = 0.0f;
#pragma unroll
  for (int d = 0; d < ND; d += 4) {
    float4 t = *(const float4*)(xr + d);
    s += t.x * t.x + t.y * t.y + t.z * t.z + t.w * t.w;
    hr[d]     = (_Float16)t.x;  hr[d + 1] = (_Float16)t.y;
    hr[d + 2] = (_Float16)t.z;  hr[d + 3] = (_Float16)t.w;
  }
  sq[row] = s;
}

// ------------- K0b: pre-swizzle w1/w2 to WMMA B-fragment layout, zero stats ----
__global__ __launch_bounds__(256) void k0b_pack(const float* __restrict__ w1,
                                                const float* __restrict__ w2,
                                                _Float16* __restrict__ w1p,
                                                _Float16* __restrict__ w2p,
                                                float* __restrict__ stats) {
  int tid = threadIdx.x;
  if (tid < 192) stats[tid] = 0.0f;
  { // w1: (128,32) -> 2 col-tiles x 4 K-slices x 32 lanes x 16 elems
    int ct = tid >> 7, kk = (tid >> 5) & 3, L = tid & 31;
    int hi = (L >> 4) & 1, col = ct * 16 + (L & 15);
#pragma unroll
    for (int e = 0; e < 16; ++e) {
      int k = kk * 32 + ((e < 8) ? (hi * 8 + e) : (16 + hi * 8 + (e - 8)));
      w1p[(size_t)tid * 16 + e] = (_Float16)w1[k * 32 + col];
    }
  }
  if (tid < 128) { // w2: (32,64) -> 4 col-tiles x 32 lanes x 16 elems
    int ct = tid >> 5, L = tid & 31;
    int hi = (L >> 4) & 1, col = ct * 16 + (L & 15);
#pragma unroll
    for (int e = 0; e < 16; ++e) {
      int k = (e < 8) ? (hi * 8 + e) : (16 + hi * 8 + (e - 8));
      w2p[(size_t)tid * 16 + e] = (_Float16)w2[k * 64 + col];
    }
  }
}

// ------------- K1: fused distance WMMA GEMM + topK + edge-feature max ----------
// grid (32,8): blockIdx.x = 128-row tile, blockIdx.y = batch. 256 thr = 8 waves.
// Candidate chunks double-buffered in LDS, filled by async-to-LDS DMA.
__global__ __launch_bounds__(256) void k1_knn_edge(const float* __restrict__ x,
                                                   const _Float16* __restrict__ xh,
                                                   const float* __restrict__ sq,
                                                   _Float16* __restrict__ feat) {
  __shared__ _Float16 lsX[2][128 * ND];    // candidate chunks (f16), double buffered
  __shared__ float    lsSq[2][128];
  __shared__ float    lsStage[8][16][16];  // per-wave distance tile
  __shared__ float    lsKV[128][NK];
  __shared__ int      lsKI[128][NK];

  const int b = blockIdx.y;
  const int rowBase = blockIdx.x * 128;
  const int tid = threadIdx.x;
  const int wave = tid >> 5, lane = tid & 31;
  const int hi = (lane >> 4) & 1, l16 = lane & 15;

  for (int i = tid; i < 128 * NK; i += 256) { lsKV[i / NK][i % NK] = FINF; lsKI[i / NK][i % NK] = 0; }

  // A fragments: 16 query rows x K=64 (two 16x32 slices)
  const int myRow = rowBase + wave * 16 + l16;
  const _Float16* arow = xh + ((size_t)b * NN + myRow) * ND;
  v16h a0 = mk16(ld8(arow + hi * 8),      ld8(arow + 16 + hi * 8));
  v16h a1 = mk16(ld8(arow + 32 + hi * 8), ld8(arow + 48 + hi * 8));

  auto issue = [&](int chunk, int buf) {  // 256 thr x 4 x 16B = 16 KB chunk
    const _Float16* src = xh + ((size_t)b * NN + chunk * 128) * ND + tid * 32;
    _Float16* dst = &lsX[buf][tid * 32];
    async_b128(dst,      src);
    async_b128(dst + 8,  src + 8);
    async_b128(dst + 16, src + 16);
    async_b128(dst + 24, src + 24);
    if (tid < 128) lsSq[buf][tid] = sq[b * NN + chunk * 128 + tid];
  };

  float worstV = FINF; int worstP = 0;     // owner lanes (lane<16): row wave*16+lane
  issue(0, 0);

  for (int chunk = 0; chunk < NN / 128; ++chunk) {
    const int cur = chunk & 1;
    const int cBase = chunk * 128;
    wait_async0();        // our part of chunk `chunk` has landed in LDS
    __syncthreads();      // whole chunk visible; prior reads of other buffer done
    if (chunk + 1 < NN / 128) issue(chunk + 1, cur ^ 1);   // overlap next DMA

    for (int jt = 0; jt < 8; ++jt) {
      const int jl = jt * 16 + l16;
      const _Float16* colp = &lsX[cur][jl * ND];
      v16h b0 = mk16(ld8(colp + hi * 8),      ld8(colp + 16 + hi * 8));
      v16h b1 = mk16(ld8(colp + 32 + hi * 8), ld8(colp + 48 + hi * 8));
      v8f c = {};
      c = wmma16(a0, b0, c);
      c = wmma16(a1, b1, c);
      const float sqj = lsSq[cur][jl];
#pragma unroll
      for (int v = 0; v < 8; ++v)           // rank key: sq_j - 2*<xi,xj>
        lsStage[wave][v + hi * 8][l16] = sqj - 2.0f * c[v];
      lds_fence();
      if (lane < 16) {
        const float4 s0 = *(const float4*)&lsStage[wave][lane][0];
        const float4 s1 = *(const float4*)&lsStage[wave][lane][4];
        const float4 s2 = *(const float4*)&lsStage[wave][lane][8];
        const float4 s3 = *(const float4*)&lsStage[wave][lane][12];
        // branch-free tile-min gate (v_min tree, dual-issuable)
        const float m0 = fminf(fminf(s0.x, s0.y), fminf(s0.z, s0.w));
        const float m1 = fminf(fminf(s1.x, s1.y), fminf(s1.z, s1.w));
        const float m2 = fminf(fminf(s2.x, s2.y), fminf(s2.z, s2.w));
        const float m3 = fminf(fminf(s3.x, s3.y), fminf(s3.z, s3.w));
        const float tmin = fminf(fminf(m0, m1), fminf(m2, m3));
        if (tmin < worstV) {                 // rare after warm-up: insert from regs
          const float s[16] = { s0.x, s0.y, s0.z, s0.w, s1.x, s1.y, s1.z, s1.w,
                                s2.x, s2.y, s2.z, s2.w, s3.x, s3.y, s3.z, s3.w };
          float* kv = lsKV[wave * 16 + lane];
          int*   ki = lsKI[wave * 16 + lane];
#pragma unroll
          for (int cc = 0; cc < 16; ++cc) {
            if (s[cc] < worstV) {
              kv[worstP] = s[cc]; ki[worstP] = cBase + jt * 16 + cc;
              float wv = -FINF; int wp = 0;
#pragma unroll
              for (int t4 = 0; t4 < 5; ++t4) {     // vectorized worst rescan
                float4 q = *(const float4*)&kv[t4 * 4];
                if (q.x > wv) { wv = q.x; wp = t4 * 4; }
                if (q.y > wv) { wv = q.y; wp = t4 * 4 + 1; }
                if (q.z > wv) { wv = q.z; wp = t4 * 4 + 2; }
                if (q.w > wv) { wv = q.w; wp = t4 * 4 + 3; }
              }
              worstV = wv; worstP = wp;
            }
          }
        }
      }
      lds_fence();
    }
  }
  __syncthreads();

  // edge features: lane -> (row l16, dim-half hi). feat[:,0:64]=x, [:,64:128]=max_k(x_j)-x
  {
    const int n = rowBase + wave * 16 + l16;
    const int* ki = lsKI[wave * 16 + l16];
    float mx[32];
#pragma unroll
    for (int d = 0; d < 32; ++d) mx[d] = -FINF;
    for (int k = 0; k < NK; ++k) {
      const float* xp = x + ((size_t)b * NN + ki[k]) * ND + hi * 32;
#pragma unroll
      for (int d = 0; d < 32; d += 4) {
        float4 t = *(const float4*)(xp + d);
        mx[d]     = fmaxf(mx[d], t.x);     mx[d + 1] = fmaxf(mx[d + 1], t.y);
        mx[d + 2] = fmaxf(mx[d + 2], t.z); mx[d + 3] = fmaxf(mx[d + 3], t.w);
      }
    }
    const float* cp = x + ((size_t)b * NN + n) * ND + hi * 32;
    _Float16* fp = feat + ((size_t)b * NN + n) * 128 + hi * 32;
#pragma unroll
    for (int d = 0; d < 32; d += 4) {
      float4 t = *(const float4*)(cp + d);
      fp[d]     = (_Float16)t.x;  fp[64 + d]     = (_Float16)(mx[d]     - t.x);
      fp[d + 1] = (_Float16)t.y;  fp[64 + d + 1] = (_Float16)(mx[d + 1] - t.y);
      fp[d + 2] = (_Float16)t.z;  fp[64 + d + 2] = (_Float16)(mx[d + 2] - t.z);
      fp[d + 3] = (_Float16)t.w;  fp[64 + d + 3] = (_Float16)(mx[d + 3] - t.w);
    }
  }
}

// ------------- K2: h1 = relu(feat @ w1 + b1), accumulate BN1 stats -------------
__global__ __launch_bounds__(256) void k2_mlp1(const _Float16* __restrict__ feat,
                                               const _Float16* __restrict__ w1p,
                                               const float* __restrict__ b1,
                                               float* __restrict__ h1,
                                               float* __restrict__ stats) {
  __shared__ float lsS[32], lsQ[32];
  const int tid = threadIdx.x;
  if (tid < 32) { lsS[tid] = 0.0f; lsQ[tid] = 0.0f; }
  __syncthreads();
  const int wave = tid >> 5, lane = tid & 31;
  const int hi = (lane >> 4) & 1, l16 = lane & 15;
  const int r0 = (blockIdx.x * 8 + wave) * 16;
  const _Float16* arow = feat + (size_t)(r0 + l16) * 128;
  v16h a[4];
#pragma unroll
  for (int kk = 0; kk < 4; ++kk)
    a[kk] = mk16(ld8(arow + kk * 32 + hi * 8), ld8(arow + kk * 32 + 16 + hi * 8));
#pragma unroll
  for (int ct = 0; ct < 2; ++ct) {
    v8f c = {};
#pragma unroll
    for (int kk = 0; kk < 4; ++kk) {
      v16h bf = *(const v16h*)(w1p + ((size_t)((ct * 4 + kk) * 32 + lane)) * 16);
      c = wmma16(a[kk], bf, c);
    }
    const int col = ct * 16 + l16;
    const float bias = b1[col];
    float s = 0.0f, q = 0.0f;
#pragma unroll
    for (int v = 0; v < 8; ++v) {
      float hv = fmaxf(c[v] + bias, 0.0f);
      h1[(size_t)(r0 + v + hi * 8) * 32 + col] = hv;
      s += hv; q += hv * hv;
    }
    atomicAdd(&lsS[col], s);
    atomicAdd(&lsQ[col], q);
  }
  __syncthreads();
  if (tid < 32) { atomicAdd(&stats[tid], lsS[tid]); atomicAdd(&stats[32 + tid], lsQ[tid]); }
}

// ------------- K3: h2 = relu(bn1(h1) @ w2 + b2), accumulate BN2 stats ----------
__global__ __launch_bounds__(256) void k3_mlp2(const float* __restrict__ h1,
                                               const _Float16* __restrict__ w2p,
                                               const float* __restrict__ b2,
                                               const float* __restrict__ g1,
                                               const float* __restrict__ be1,
                                               float* __restrict__ stats,
                                               float* __restrict__ h2) {
  __shared__ float lsS[64], lsQ[64];
  const int tid = threadIdx.x;
  if (tid < 64) { lsS[tid] = 0.0f; lsQ[tid] = 0.0f; }
  __syncthreads();
  const int wave = tid >> 5, lane = tid & 31;
  const int hi = (lane >> 4) & 1, l16 = lane & 15;
  float ns[16], nb[16];
#pragma unroll
  for (int e = 0; e < 16; ++e) {
    int k = (e < 8) ? (hi * 8 + e) : (16 + hi * 8 + (e - 8));
    float mean = stats[k] * (1.0f / TOTROWS);
    float var = stats[32 + k] * (1.0f / TOTROWS) - mean * mean;
    float sc = g1[k] * rsqrtf(var + BN_EPS);
    ns[e] = sc; nb[e] = be1[k] - mean * sc;
  }
  const int r0 = (blockIdx.x * 8 + wave) * 16;
  const float* hrow = h1 + (size_t)(r0 + l16) * 32;
  v16h a;
#pragma unroll
  for (int e = 0; e < 16; ++e) {
    int k = (e < 8) ? (hi * 8 + e) : (16 + hi * 8 + (e - 8));
    a[e] = (_Float16)(hrow[k] * ns[e] + nb[e]);
  }
#pragma unroll
  for (int ct = 0; ct < 4; ++ct) {
    v16h bf = *(const v16h*)(w2p + (size_t)(ct * 32 + lane) * 16);
    v8f c = {};
    c = wmma16(a, bf, c);
    const int col = ct * 16 + l16;
    const float bias = b2[col];
    float s = 0.0f, q = 0.0f;
#pragma unroll
    for (int v = 0; v < 8; ++v) {
      float hv = fmaxf(c[v] + bias, 0.0f);
      h2[(size_t)(r0 + v + hi * 8) * 64 + col] = hv;
      s += hv; q += hv * hv;
    }
    atomicAdd(&lsS[col], s);
    atomicAdd(&lsQ[col], q);
  }
  __syncthreads();
  if (tid < 64) { atomicAdd(&stats[64 + tid], lsS[tid]); atomicAdd(&stats[128 + tid], lsQ[tid]); }
}

// ------------- K4: pooled = max_n bn2(h2) ------------------------------------
__global__ __launch_bounds__(256) void k4_pool(const float* __restrict__ h2,
                                               const float* __restrict__ g2,
                                               const float* __restrict__ be2,
                                               const float* __restrict__ stats,
                                               float* __restrict__ pooled) {
  __shared__ float red[256];
  const int b = blockIdx.x, tid = threadIdx.x;
  const int ch = tid & 63, grp = tid >> 6;
  float mean = stats[64 + ch] * (1.0f / TOTROWS);
  float var = stats[128 + ch] * (1.0f / TOTROWS) - mean * mean;
  float sc = g2[ch] * rsqrtf(var + BN_EPS);
  float bi = be2[ch] - mean * sc;
  float mx = -FINF;
  for (int n = grp; n < NN; n += 4)
    mx = fmaxf(mx, h2[((size_t)b * NN + n) * 64 + ch] * sc + bi);
  red[tid] = mx;
  __syncthreads();
  if (grp == 0)
    pooled[b * 64 + ch] = fmaxf(fmaxf(red[ch], red[64 + ch]), fmaxf(red[128 + ch], red[192 + ch]));
}

// ------------- K5: out = softmax(pooled @ wd + bd) ---------------------------
__global__ __launch_bounds__(256) void k5_decode(const float* __restrict__ pooled,
                                                 const float* __restrict__ wd,
                                                 const float* __restrict__ bd,
                                                 float* __restrict__ out) {
  __shared__ float lg[NN];
  __shared__ float p[64];
  __shared__ float red[256];
  const int b = blockIdx.x, tid = threadIdx.x;
  if (tid < 64) p[tid] = pooled[b * 64 + tid];
  __syncthreads();
  float lmax = -FINF;
  for (int j = tid; j < NN; j += 256) {
    float acc = bd[j];
#pragma unroll
    for (int d = 0; d < 64; ++d) acc += p[d] * wd[(size_t)d * NN + j];
    lg[j] = acc;
    lmax = fmaxf(lmax, acc);
  }
  red[tid] = lmax;
  __syncthreads();
  for (int s = 128; s > 0; s >>= 1) { if (tid < s) red[tid] = fmaxf(red[tid], red[tid + s]); __syncthreads(); }
  const float gmax = red[0];
  __syncthreads();
  float lsum = 0.0f;
  for (int j = tid; j < NN; j += 256) { float e = __expf(lg[j] - gmax); lg[j] = e; lsum += e; }
  red[tid] = lsum;
  __syncthreads();
  for (int s = 128; s > 0; s >>= 1) { if (tid < s) red[tid] += red[tid + s]; __syncthreads(); }
  const float inv = 1.0f / red[0];
  for (int j = tid; j < NN; j += 256) out[(size_t)b * NN + j] = lg[j] * inv;
}

// ---------------- workspace layout (peak ~20 MB) -----------------------------
static constexpr size_t OFF_XH   = 0;                         // 32768*64*2  = 4 MiB
static constexpr size_t OFF_SQ   = (size_t)4 << 20;           // 128 KiB
static constexpr size_t OFF_W1P  = OFF_SQ + ((size_t)128 << 10);
static constexpr size_t OFF_W2P  = OFF_W1P + ((size_t)8 << 10);
static constexpr size_t OFF_ST   = OFF_W2P + ((size_t)4 << 10); // 192 floats
static constexpr size_t OFF_PL   = OFF_ST + 1024;             // pooled 512 f
static constexpr size_t OFF_FEAT = (size_t)8 << 20;           // 8 MiB (f16)
static constexpr size_t OFF_H2   = (size_t)8 << 20;           // aliases FEAT (dead after K2)
static constexpr size_t OFF_H1   = (size_t)16 << 20;          // 4 MiB

extern "C" void kernel_launch(void* const* d_in, const int* in_sizes, int n_in,
                              void* d_out, int out_size, void* d_ws, size_t ws_size,
                              hipStream_t stream) {
  (void)in_sizes; (void)n_in; (void)out_size; (void)ws_size;
  const float* x   = (const float*)d_in[0];
  const float* w1  = (const float*)d_in[1];
  const float* b1  = (const float*)d_in[2];
  const float* g1  = (const float*)d_in[3];
  const float* be1 = (const float*)d_in[4];
  const float* w2  = (const float*)d_in[5];
  const float* b2  = (const float*)d_in[6];
  const float* g2  = (const float*)d_in[7];
  const float* be2 = (const float*)d_in[8];
  const float* wd  = (const float*)d_in[9];
  const float* bd  = (const float*)d_in[10];
  float* out = (float*)d_out;

  char* ws = (char*)d_ws;
  _Float16* xh    = (_Float16*)(ws + OFF_XH);
  float*    sq    = (float*)(ws + OFF_SQ);
  _Float16* w1p   = (_Float16*)(ws + OFF_W1P);
  _Float16* w2p   = (_Float16*)(ws + OFF_W2P);
  float*    stats = (float*)(ws + OFF_ST);
  float*    pool  = (float*)(ws + OFF_PL);
  _Float16* feat  = (_Float16*)(ws + OFF_FEAT);
  float*    h1    = (float*)(ws + OFF_H1);
  float*    h2    = (float*)(ws + OFF_H2);

  k0a_convert<<<TOTROWS / 256, 256, 0, stream>>>(x, xh, sq);
  k0b_pack<<<1, 256, 0, stream>>>(w1, w2, w1p, w2p, stats);
  k1_knn_edge<<<dim3(NN / 128, NB), 256, 0, stream>>>(x, xh, sq, feat);
  k2_mlp1<<<TOTROWS / 128, 256, 0, stream>>>(feat, w1p, b1, h1, stats);
  k3_mlp2<<<TOTROWS / 128, 256, 0, stream>>>(h1, w2p, b2, g1, be1, stats, h2);
  k4_pool<<<NB, 256, 0, stream>>>(h2, g2, be2, stats, pool);
  k5_decode<<<NB, 256, 0, stream>>>(pool, wd, bd, out);
}
